// Block_34557306864055
// MI455X (gfx1250) — matrix operand order repaired and checked by reading, not compiled, not run on hardware
//
#include <hip/hip_runtime.h>

// ---------------------------------------------------------------------------
// Transformer block for MI455X (gfx1250, wave32, WMMA bf16 16x16x32, f32 acc)
// B=4, N=2048, C=768, H=12, D=64, HID=3072
// Round 3: fix async builtin pointer types (int4 vec ptrs w/ address spaces).
// ---------------------------------------------------------------------------

typedef __attribute__((ext_vector_type(16))) __bf16 v16bf;
typedef __attribute__((ext_vector_type(8)))  __bf16 v8bf;
typedef __attribute__((ext_vector_type(8)))  float  v8f;

union BF16x16 { v16bf v; v8bf h[2]; };

__device__ __forceinline__ unsigned short f2bf(float f) {
    union { float f; unsigned int u; } cv; cv.f = f;
    unsigned int r = cv.u + 0x7FFFu + ((cv.u >> 16) & 1u);   // round-nearest-even
    return (unsigned short)(r >> 16);
}

#define SEQ   2048
#define CDIM  768
#define NHEAD 12
#define DHEAD 64
#define S3C   2304
#define MROWS 8192            // B*SEQ

// ----------------------- async global -> LDS plumbing ----------------------
#if defined(__has_builtin)
#if __has_builtin(__builtin_amdgcn_global_load_async_to_lds_b128)
#define HAVE_ASYNC_LDS 1
#endif
#endif

#ifdef HAVE_ASYNC_LDS
typedef __attribute__((ext_vector_type(4))) int int4v;
typedef __attribute__((address_space(1))) int4v as1_int4;
typedef __attribute__((address_space(3))) int4v as3_int4;
__device__ __forceinline__ void async_cp16(const void* g, void* l) {
    __builtin_amdgcn_global_load_async_to_lds_b128((as1_int4*)g, (as3_int4*)l, 0, 0);
}
#if __has_builtin(__builtin_amdgcn_s_wait_asynccnt)
__device__ __forceinline__ void wait_async0() { __builtin_amdgcn_s_wait_asynccnt(0); }
__device__ __forceinline__ void wait_async4() { __builtin_amdgcn_s_wait_asynccnt(4); }
#else
__device__ __forceinline__ void wait_async0() { asm volatile("s_wait_asynccnt 0x0" ::: "memory"); }
__device__ __forceinline__ void wait_async4() { asm volatile("s_wait_asynccnt 0x4" ::: "memory"); }
#endif
#else
// fallback: synchronous copy through VGPRs (round-1 behavior)
__device__ __forceinline__ void async_cp16(const void* g, void* l) {
    *(uint4*)l = *(const uint4*)g;
}
__device__ __forceinline__ void wait_async0() {}
__device__ __forceinline__ void wait_async4() {}
#endif

// =========================== fp32 -> bf16 convert ==========================
__global__ void cvt_bf16_kernel(const float* __restrict__ src,
                                unsigned short* __restrict__ dst, int n) {
    int i = blockIdx.x * blockDim.x + threadIdx.x;
    if (i < n) dst[i] = f2bf(src[i]);
}

// ================================ LayerNorm ================================
__global__ __launch_bounds__(256) void ln_bf16_kernel(
    const float* __restrict__ x, const float* __restrict__ g,
    const float* __restrict__ bb, unsigned short* __restrict__ y) {
    const int t = threadIdx.x;
    const size_t row = blockIdx.x;
    const float* xr = x + row * CDIM;
    float v0 = xr[t], v1 = xr[t + 256], v2 = xr[t + 512];
    __shared__ float2 red[256];
    red[t] = make_float2(v0 + v1 + v2, v0 * v0 + v1 * v1 + v2 * v2);
    __syncthreads();
    for (int s = 128; s > 0; s >>= 1) {
        if (t < s) {
            float2 a = red[t], b2 = red[t + s];
            red[t] = make_float2(a.x + b2.x, a.y + b2.y);
        }
        __syncthreads();
    }
    const float mu  = red[0].x * (1.0f / CDIM);
    const float var = red[0].y * (1.0f / CDIM) - mu * mu;
    const float rst = rsqrtf(var + 1e-5f);
    unsigned short* yr = y + row * CDIM;
    yr[t]       = f2bf((v0 - mu) * rst * g[t]       + bb[t]);
    yr[t + 256] = f2bf((v1 - mu) * rst * g[t + 256] + bb[t + 256]);
    yr[t + 512] = f2bf((v2 - mu) * rst * g[t + 512] + bb[t + 512]);
}

// ================================== GEMM ===================================
// out[m,n] = sum_k A[m,k] * W[n,k]  (A: [M,K] bf16, W: [Nf,K] bf16)
// WG 256 thr = 8 waves (2x4), WG tile 128x128, wave tile 64x32, BK=32.
// Double-buffered LDS tiles filled by async global->LDS copies.
// EPI: 0 = bf16 out; 1 = bf16 out, cols<768 scaled by 1/8 (QKV, q*D^-0.5);
//      2 = +bias, exact GELU, bf16 out; 3 = +bias +resid(f32), f32 out.
#define LDSS 40   // padded row stride (bf16 elems) -> conflict-free frag reads

template <int EPI>
__global__ __launch_bounds__(256) void gemm_bf16_kernel(
    const unsigned short* __restrict__ A, const unsigned short* __restrict__ W,
    const float* __restrict__ bias, const float* __restrict__ resid,
    unsigned short* __restrict__ outb, float* __restrict__ outf,
    int M, int K, int Nf) {
    __shared__ unsigned short As[2][128][LDSS];
    __shared__ unsigned short Bs[2][128][LDSS];

    const int t = threadIdx.x;
    const int lane = t & 31, wave = t >> 5;
    const int wm = wave >> 2, wn = wave & 3;       // 2 x 4 wave grid
    const int lm = lane & 15, lh = lane >> 4;
    const int m0 = blockIdx.x * 128;
    const int n0 = blockIdx.y * 128;

    v8f acc[4][2];
#pragma unroll
    for (int i = 0; i < 4; ++i)
#pragma unroll
        for (int j = 0; j < 2; ++j) acc[i][j] = (v8f){};

    auto load_batch = [&](int kt) {                // 512 16B-chunks / 256 thr
        const int buf = kt & 1, k0 = kt * 32;
#pragma unroll
        for (int i = 0; i < 2; ++i) {
            int chunk = t + i * 256;
            int row = chunk >> 2, c = (chunk & 3) * 8;
            async_cp16(A + (size_t)(m0 + row) * K + k0 + c, &As[buf][row][c]);
            async_cp16(W + (size_t)(n0 + row) * K + k0 + c, &Bs[buf][row][c]);
        }
    };

    const int nk = K >> 5;
    load_batch(0);
    for (int kt = 0; kt < nk; ++kt) {
        if (kt + 1 < nk) {
            load_batch(kt + 1);
            wait_async4();                          // batch kt landed
        } else {
            wait_async0();
        }
        __syncthreads();

        const int buf = kt & 1;
        BF16x16 af[4], bfr[2];
#pragma unroll
        for (int i = 0; i < 4; ++i) {              // A frags: lane = row
            int row = wm * 64 + i * 16 + lm;
            af[i].h[0] = *(const v8bf*)&As[buf][row][lh * 8];
            af[i].h[1] = *(const v8bf*)&As[buf][row][16 + lh * 8];
        }
#pragma unroll
        for (int j = 0; j < 2; ++j) {              // B frags: lane = col (W row)
            int row = wn * 32 + j * 16 + lm;
            bfr[j].h[0] = *(const v8bf*)&Bs[buf][row][lh * 8];
            bfr[j].h[1] = *(const v8bf*)&Bs[buf][row][16 + lh * 8];
        }
#pragma unroll
        for (int i = 0; i < 4; ++i)
#pragma unroll
            for (int j = 0; j < 2; ++j)
                acc[i][j] = __builtin_amdgcn_wmma_f32_16x16x32_bf16(
                    false, af[i].v, false, bfr[j].v, (short)0, acc[i][j], false, false);
        __syncthreads();
    }

#pragma unroll
    for (int i = 0; i < 4; ++i)
#pragma unroll
        for (int j = 0; j < 2; ++j) {
            const int gcol = n0 + wn * 32 + j * 16 + lm;
#pragma unroll
            for (int r = 0; r < 8; ++r) {
                const int grow = m0 + wm * 64 + i * 16 + r + 8 * lh;
                float v = acc[i][j][r];
                size_t idx = (size_t)grow * Nf + gcol;
                if (EPI == 0) {
                    outb[idx] = f2bf(v);
                } else if (EPI == 1) {
                    if (gcol < 768) v *= 0.125f;   // q * D^-0.5
                    outb[idx] = f2bf(v);
                } else if (EPI == 2) {
                    v += bias[gcol];
                    v = 0.5f * v * (1.0f + erff(v * 0.70710678118654752f));
                    outb[idx] = f2bf(v);
                } else {                            // EPI == 3
                    v += bias[gcol] + resid[idx];
                    outf[idx] = v;
                }
            }
        }
}

// ============================ Flash attention ==============================
// One WG (128 thr = 4 waves) per (b, h, 64-query block). Each wave owns 16
// query rows. K chunk (async global->LDS) and V^T chunk staged in LDS; online
// softmax; P goes through LDS to reach A-layout for the P*V WMMA.
__global__ __launch_bounds__(128) void attn_kernel(
    const unsigned short* __restrict__ qkv, unsigned short* __restrict__ obf) {
    __shared__ unsigned short Ks [64][72];
    __shared__ unsigned short Vts[64][72];
    __shared__ unsigned short Ps [64][72];

    const int t = threadIdx.x;
    const int lane = t & 31, wave = t >> 5;
    const int lm = lane & 15, lh = lane >> 4;
    const int wg = blockIdx.x;
    const int qb = wg & 31;                 // 2048/64 query blocks
    const int bh = wg >> 5;
    const int b  = bh / NHEAD, h = bh % NHEAD;
    const size_t rowbase = (size_t)b * SEQ;
    const int m0   = qb * 64;
    const int qcol = h * DHEAD;
    const int kcol = 768 + h * DHEAD;
    const int vcol = 1536 + h * DHEAD;

    // Q A-frags for this wave's 16 rows (D=64 -> two k-chunks of 32)
    BF16x16 aQ[2];
    {
        const unsigned short* qp =
            qkv + (rowbase + m0 + wave * 16 + lm) * S3C + qcol;
#pragma unroll
        for (int kk = 0; kk < 2; ++kk) {
            aQ[kk].h[0] = *(const v8bf*)(qp + kk * 32 + lh * 8);
            aQ[kk].h[1] = *(const v8bf*)(qp + kk * 32 + 16 + lh * 8);
        }
    }

    float mrow[8], lrow[8];
    v8f accO[4];
#pragma unroll
    for (int r = 0; r < 8; ++r) { mrow[r] = -1e30f; lrow[r] = 0.0f; }
#pragma unroll
    for (int nt = 0; nt < 4; ++nt) accO[nt] = (v8f){};

    for (int j0 = 0; j0 < SEQ; j0 += 64) {
        // ---- stage K chunk (async) ----
#pragma unroll
        for (int i = 0; i < 4; ++i) {          // 512 chunks of 16B
            int chunk = t + i * 128;
            int r = chunk >> 3, c = (chunk & 7) * 8;
            async_cp16(qkv + (rowbase + j0 + r) * S3C + kcol + c, &Ks[r][c]);
        }
        // ---- stage V^T chunk (transpose on LDS store) ----
#pragma unroll
        for (int i = 0; i < 4; ++i) {
            int chunk = t + i * 128;
            int key = chunk & 63, d0 = (chunk >> 6) * 8;
            uint4 raw = *(const uint4*)(qkv + (rowbase + j0 + key) * S3C + vcol + d0);
            const unsigned short* pv = (const unsigned short*)&raw;
#pragma unroll
            for (int e = 0; e < 8; ++e) Vts[d0 + e][key] = pv[e];
        }
        wait_async0();
        __syncthreads();

        // ---- S = Q * K^T  (4 tiles of 16x16 keys) ----
        v8f S[4];
#pragma unroll
        for (int nt = 0; nt < 4; ++nt) {
            v8f s = (v8f){};
#pragma unroll
            for (int kk = 0; kk < 2; ++kk) {
                BF16x16 bK;
                const unsigned short* kp = &Ks[nt * 16 + lm][kk * 32 + lh * 8];
                bK.h[0] = *(const v8bf*)kp;
                bK.h[1] = *(const v8bf*)(kp + 16);
                s = __builtin_amdgcn_wmma_f32_16x16x32_bf16(
                    false, aQ[kk].v, false, bK.v, (short)0, s, false, false);
            }
            S[nt] = s;
        }

        // ---- online softmax (row reductions across 16-lane groups) ----
#pragma unroll
        for (int r = 0; r < 8; ++r) {
            float mx = S[0][r];
#pragma unroll
            for (int nt = 1; nt < 4; ++nt) mx = fmaxf(mx, S[nt][r]);
#pragma unroll
            for (int msk = 1; msk < 16; msk <<= 1)
                mx = fmaxf(mx, __shfl_xor(mx, msk, 16));
            float mnew  = fmaxf(mrow[r], mx);
            float alpha = __expf(mrow[r] - mnew);
            mrow[r] = mnew;
            float rs = 0.0f;
#pragma unroll
            for (int nt = 0; nt < 4; ++nt) {
                float p = __expf(S[nt][r] - mnew);
                rs += p;
                Ps[wave * 16 + r + 8 * lh][nt * 16 + lm] = f2bf(p);
            }
#pragma unroll
            for (int msk = 1; msk < 16; msk <<= 1)
                rs += __shfl_xor(rs, msk, 16);
            lrow[r] = lrow[r] * alpha + rs;
#pragma unroll
            for (int nt = 0; nt < 4; ++nt) accO[nt][r] *= alpha;
        }
        __syncthreads();

        // ---- O += P * V  (B-frags from V^T rows = output-d columns) ----
        BF16x16 aP[2];
        {
            const unsigned short* pp = &Ps[wave * 16 + lm][0];
#pragma unroll
            for (int kk = 0; kk < 2; ++kk) {
                aP[kk].h[0] = *(const v8bf*)(pp + kk * 32 + lh * 8);
                aP[kk].h[1] = *(const v8bf*)(pp + kk * 32 + 16 + lh * 8);
            }
        }
#pragma unroll
        for (int nt = 0; nt < 4; ++nt)
#pragma unroll
            for (int kk = 0; kk < 2; ++kk) {
                BF16x16 bV;
                const unsigned short* vp = &Vts[nt * 16 + lm][kk * 32 + lh * 8];
                bV.h[0] = *(const v8bf*)vp;
                bV.h[1] = *(const v8bf*)(vp + 16);
                accO[nt] = __builtin_amdgcn_wmma_f32_16x16x32_bf16(
                    false, aP[kk].v, false, bV.v, (short)0, accO[nt], false, false);
            }
        __syncthreads();
    }

    // ---- normalize + write O back in [B,N,C] layout (col = h*64+d) ----
#pragma unroll
    for (int r = 0; r < 8; ++r) {
        float inv = 1.0f / lrow[r];
        size_t grow = rowbase + m0 + wave * 16 + r + 8 * lh;
#pragma unroll
        for (int nt = 0; nt < 4; ++nt)
            obf[grow * CDIM + h * DHEAD + nt * 16 + lm] =
                f2bf(accO[nt][r] * inv);
    }
}

// ================================= launch ==================================
extern "C" void kernel_launch(void* const* d_in, const int* in_sizes, int n_in,
                              void* d_out, int out_size, void* d_ws, size_t ws_size,
                              hipStream_t stream) {
    const float* x      = (const float*)d_in[0];
    const float* ln1_g  = (const float*)d_in[1];
    const float* ln1_b  = (const float*)d_in[2];
    const float* qkv_w  = (const float*)d_in[3];
    const float* proj_w = (const float*)d_in[4];
    const float* proj_b = (const float*)d_in[5];
    const float* ln2_g  = (const float*)d_in[6];
    const float* ln2_b  = (const float*)d_in[7];
    const float* fc1_w  = (const float*)d_in[8];
    const float* fc1_b  = (const float*)d_in[9];
    const float* fc2_w  = (const float*)d_in[10];
    const float* fc2_b  = (const float*)d_in[11];
    float* out = (float*)d_out;

    char* ws = (char*)d_ws;
    size_t off = 0;
    auto alloc = [&](size_t bytes) { char* p = ws + off; off += (bytes + 255) & ~(size_t)255; return p; };
    unsigned short* wq   = (unsigned short*)alloc((size_t)S3C * CDIM * 2);
    unsigned short* wp   = (unsigned short*)alloc((size_t)CDIM * CDIM * 2);
    unsigned short* w1   = (unsigned short*)alloc((size_t)3072 * CDIM * 2);
    unsigned short* w2   = (unsigned short*)alloc((size_t)CDIM * 3072 * 2);
    unsigned short* hbf  = (unsigned short*)alloc((size_t)MROWS * CDIM * 2);   // LN1/LN2 out
    unsigned short* qkvb = (unsigned short*)alloc((size_t)MROWS * S3C * 2);
    unsigned short* obf  = (unsigned short*)alloc((size_t)MROWS * CDIM * 2);
    unsigned short* fco  = (unsigned short*)alloc((size_t)MROWS * 3072 * 2);

    // weights -> bf16
    cvt_bf16_kernel<<<(S3C * CDIM + 255) / 256, 256, 0, stream>>>(qkv_w,  wq, S3C * CDIM);
    cvt_bf16_kernel<<<(CDIM * CDIM + 255) / 256, 256, 0, stream>>>(proj_w, wp, CDIM * CDIM);
    cvt_bf16_kernel<<<(3072 * CDIM + 255) / 256, 256, 0, stream>>>(fc1_w,  w1, 3072 * CDIM);
    cvt_bf16_kernel<<<(CDIM * 3072 + 255) / 256, 256, 0, stream>>>(fc2_w,  w2, CDIM * 3072);

    // LN1
    ln_bf16_kernel<<<MROWS, 256, 0, stream>>>(x, ln1_g, ln1_b, hbf);
    // QKV projection (q pre-scaled by D^-0.5)
    gemm_bf16_kernel<1><<<dim3(MROWS / 128, S3C / 128), 256, 0, stream>>>(
        hbf, wq, nullptr, nullptr, qkvb, nullptr, MROWS, CDIM, S3C);
    // attention
    attn_kernel<<<4 * NHEAD * (SEQ / 64), 128, 0, stream>>>(qkvb, obf);
    // output projection + bias + residual (x) -> d_out (f32)
    gemm_bf16_kernel<3><<<dim3(MROWS / 128, CDIM / 128), 256, 0, stream>>>(
        obf, wp, proj_b, x, nullptr, out, MROWS, CDIM, CDIM);
    // LN2 on residual stream
    ln_bf16_kernel<<<MROWS, 256, 0, stream>>>(out, ln2_g, ln2_b, hbf);
    // FC1 + bias + exact GELU
    gemm_bf16_kernel<2><<<dim3(MROWS / 128, 3072 / 128), 256, 0, stream>>>(
        hbf, w1, fc1_b, nullptr, fco, nullptr, MROWS, CDIM, 3072);
    // FC2 + bias + residual (in-place on d_out)
    gemm_bf16_kernel<3><<<dim3(MROWS / 128, CDIM / 128), 256, 0, stream>>>(
        fco, w2, fc2_b, out, nullptr, out, MROWS, 3072, CDIM);
}
